// FeatureMerge_29764123361765
// MI455X (gfx1250) — matrix operand order repaired and checked by reading, not compiled, not run on hardware
//
#include <hip/hip_runtime.h>
#include <hip/hip_bf16.h>

// ---- problem constants (from reference) ----
#define H      768          // hidden
#define K2     1536         // 2*H (GEMM K)
#define NL     64           // num labels
// ---- tiling (async kernel) ----
#define BM     128          // rows per block
#define BN     128          // cols per block
#define BK     32           // K per step (bf16 WMMA K)
#define ASTR   144          // A tile row stride in bytes (32 f32 + pad, 16B mult, bank-spread)
#define BSTR   80           // B tile col stride in bytes (32 bf16 + pad, 16B mult, bank-spread)

typedef __attribute__((ext_vector_type(16))) __bf16 bf16x16;
typedef __attribute__((ext_vector_type(4)))  __bf16 bf16x4;
typedef __attribute__((ext_vector_type(8)))  float  floatx8;
typedef __attribute__((ext_vector_type(4)))  float  f4_t;    // plain vector (AS-copyable)
typedef __attribute__((ext_vector_type(4)))  int    v4i_t;

typedef __attribute__((address_space(1))) v4i_t* gv4p;          // global int4*
typedef __attribute__((address_space(3))) v4i_t* lv4p;          // LDS int4*
typedef __attribute__((address_space(3))) void*  as3_vp;
typedef const __attribute__((address_space(1))) f4_t* gf4p;     // global float4*

static __device__ __forceinline__ bf16x4 cvt4(f4_t v) {
    bf16x4 r;
    r.x = (__bf16)v.x; r.y = (__bf16)v.y; r.z = (__bf16)v.z; r.w = (__bf16)v.w;
    return r;
}

// fp32 -> bf16 RNE, packed pair (used by fallback kernel)
static __device__ __forceinline__ unsigned f2bf_pk(float lo, float hi) {
    union { float f; unsigned u; } a, b;
    a.f = lo; b.f = hi;
    unsigned ua = (a.u + 0x7FFFu + ((a.u >> 16) & 1u)) >> 16;
    unsigned ub = (b.u + 0x7FFFu + ((b.u >> 16) & 1u)) >> 16;
    return (ua & 0xFFFFu) | (ub << 16);
}

// async global -> LDS copy, 16 bytes per lane (ASYNCcnt-tracked)
static __device__ __forceinline__ void async_copy_b128(const void* g, void* l) {
#if __has_builtin(__builtin_amdgcn_global_load_async_to_lds_b128)
    __builtin_amdgcn_global_load_async_to_lds_b128(
        (gv4p)(__SIZE_TYPE__)g,          // AS1 v4i*  (via integer: same bit pattern)
        (lv4p)(v4i_t*)l,                 // generic -> AS3 v4i* addrspacecast
        0, 0);
#else
    unsigned lofs = (unsigned)(__SIZE_TYPE__)(as3_vp)l;
    asm volatile("global_load_async_to_lds_b128 %0, %1, off"
                 :: "v"(lofs),
                    "v"((unsigned long long)(__SIZE_TYPE__)g)
                 : "memory");
#endif
}

static __device__ __forceinline__ void wait_asynccnt0() {
#if __has_builtin(__builtin_amdgcn_s_wait_asynccnt)
    __builtin_amdgcn_s_wait_asynccnt(0);
#else
    asm volatile("s_wait_asynccnt 0x0" ::: "memory");
#endif
}

// ---------------------------------------------------------------------------
// Pass 1: convert merge_w (fp32 [H, 2H]) -> bf16 into workspace
// ---------------------------------------------------------------------------
__global__ void __launch_bounds__(256)
fm_convert_w(const float* __restrict__ W, unsigned short* __restrict__ Wbf, int n) {
    int i = (blockIdx.x * 256 + threadIdx.x) * 4;
    if (i + 3 < n) {
        f4_t v = *(gf4p)(W + i);
        *(bf16x4*)(Wbf + i) = cvt4(v);
    }
}

// ---------------------------------------------------------------------------
// Pass 2: gather + GEMM + bias + select.
//   - A and B tiles staged to LDS purely with global_load_async_to_lds_b128
//     (A as raw fp32 incl. the label gather; B as pre-converted bf16).
//   - Double-buffered LDS, padded strides (bank-conflict free fragment loads).
//   - Each wave: 32 rows x 64 cols -> 8 v_wmma_f32_16x16x32_bf16 per k-step.
// ---------------------------------------------------------------------------
__global__ void __launch_bounds__(256)
fm_gemm_async(const float* __restrict__ tokens,            // [N, H] f32
              const int*   __restrict__ labels,            // [N]
              const float* __restrict__ lfeat,             // [NL, H] f32
              const unsigned short* __restrict__ Wbf,      // [H, 2H] bf16
              const float* __restrict__ bias,              // [H] f32
              float*       __restrict__ out,               // [N, H] f32
              int n_tokens)
{
    __shared__ __align__(16) char ash[2][BM * ASTR];   // A tiles: f32 [row][k], padded
    __shared__ __align__(16) char bsh[2][BN * BSTR];   // B tiles: bf16 [col][k], padded
    __shared__ int labs[BM];

    const int tid     = threadIdx.x;
    const int rowBase = blockIdx.y * BM;
    const int colBase = blockIdx.x * BN;

    if (tid < BM) {
        int n = rowBase + tid;
        labs[tid] = (n < n_tokens) ? labels[n] : 0;
    }
    __syncthreads();

    // ---- per-thread A async-fill addressing (4 x 16B per step) ----
    // thread t handles rows r0+32*it (it=0..3), 16B chunk 'chunk' of the 128B k-slice
    const int chunk = tid & 7;
    const int r0    = tid >> 3;                        // 0..31
    unsigned curOff[4], cOff[4];                       // byte offsets into tokens / lfeat
    #pragma unroll
    for (int it = 0; it < 4; ++it) {
        int row = r0 + 32 * it;
        int n   = rowBase + row; if (n >= n_tokens) n = n_tokens - 1;
        curOff[it] = (unsigned)n * (H * 4) + chunk * 16;
        int id = labs[row] - 1;
        if (id < 0) id = 0; if (id > NL - 1) id = NL - 1;
        cOff[it] = (unsigned)id * (H * 4) + chunk * 16;
    }
    const char* curBase = (const char*)tokens;
    const unsigned aDstOff = (unsigned)r0 * ASTR + chunk * 16;   // +it*32*ASTR

    // ---- per-thread B async-fill addressing (2 x 16B per step) ----
    // tasks: 128 cols x 4 parts; thread t -> col=t>>2, part=t&3 ; second task +64 cols
    const unsigned bOff0 = (unsigned)(colBase + (tid >> 2)) * (K2 * 2) + (tid & 3) * 16;
    const unsigned bDst0 = (unsigned)(tid >> 2) * BSTR + (tid & 3) * 16;
    const char* bBase = (const char*)Wbf;

    const int w    = tid >> 5;          // wave id 0..7 (wave32)
    const int lane = tid & 31;
    const int half = lane >> 4;
    const int l15  = lane & 15;
    const int rowW = (w >> 1) * 32;     // wave's row group within block
    const int colW = (w & 1) * 64;      // wave's col group within block

    floatx8 acc[2][4] = {};             // 32x64 per wave = 8 fragments
    constexpr int STEPS = K2 / BK;      // 48

    // ---- prologue: async-fill buffer 0 for k0 = 0 ----
    #pragma unroll
    for (int it = 0; it < 4; ++it)
        async_copy_b128(curBase + curOff[it], &ash[0][aDstOff + it * (32 * ASTR)]);
    async_copy_b128(bBase + bOff0,          &bsh[0][bDst0]);
    async_copy_b128(bBase + bOff0 + 64u * (K2 * 2), &bsh[0][bDst0 + 64 * BSTR]);

    for (int i = 0; i < STEPS; ++i) {
        const int p = i & 1;

        wait_asynccnt0();               // buffer p's async fills complete (this wave)
        __syncthreads();                // all fills of p visible; reads of p^1 retired

        // ---- issue next step's async fills into buffer p^1 ----
        const int  k0n     = (i + 1) * BK;
        const bool hasNext = (i + 1 < STEPS);
        if (k0n == H) {                 // uniform, once: switch A sources to centers
            curBase = (const char*)lfeat;
            curOff[0] = cOff[0]; curOff[1] = cOff[1];
            curOff[2] = cOff[2]; curOff[3] = cOff[3];
        }
        if (hasNext) {
            const unsigned kkB = (unsigned)((k0n < H) ? k0n : (k0n - H)) * 4;
            char* aD = &ash[p ^ 1][aDstOff];
            #pragma unroll
            for (int it = 0; it < 4; ++it)
                async_copy_b128(curBase + curOff[it] + kkB, aD + it * (32 * ASTR));
            const char* bS = bBase + bOff0 + (unsigned)k0n * 2;
            char* bD = &bsh[p ^ 1][bDst0];
            async_copy_b128(bS,                    bD);
            async_copy_b128(bS + 64u * (K2 * 2),   bD + 64 * BSTR);
        }

        // ---- A fragments (2 x 16x32): fp32 from LDS + cvt -> bf16 ----
        const char* ab = &ash[p][0];
        const char* bb = &bsh[p][0];
        bf16x16 af[2];
        #pragma unroll
        for (int g = 0; g < 2; ++g) {
            int row = rowW + g * 16 + l15;
            const char* rp = ab + row * ASTR + half * 32;
            f4_t f0 = *(const f4_t*)(rp);
            f4_t f1 = *(const f4_t*)(rp + 16);
            f4_t f2 = *(const f4_t*)(rp + 64);
            f4_t f3 = *(const f4_t*)(rp + 80);
            union { bf16x16 v; bf16x4 q[4]; } ua;
            ua.q[0] = cvt4(f0); ua.q[1] = cvt4(f1);
            ua.q[2] = cvt4(f2); ua.q[3] = cvt4(f3);
            af[g] = ua.v;
        }

        // ---- B fragments (4 x 32x16), then 8 back-to-back WMMAs ----
        union { bf16x16 v; uint4 u[2]; } ub[4];
        #pragma unroll
        for (int c = 0; c < 4; ++c) {
            const char* cp = bb + (colW + c * 16 + l15) * BSTR + half * 16;
            ub[c].u[0] = *(const uint4*)(cp);
            ub[c].u[1] = *(const uint4*)(cp + 32);
        }
        #pragma unroll
        for (int g = 0; g < 2; ++g) {
            #pragma unroll
            for (int c = 0; c < 4; ++c) {
                acc[g][c] = __builtin_amdgcn_wmma_f32_16x16x32_bf16(
                    false, af[g], false, ub[c].v, (short)0, acc[g][c], false, false);
            }
        }
    }

    // ---- epilogue: bias + active-mask select + store ----
    #pragma unroll
    for (int g = 0; g < 2; ++g) {
        #pragma unroll
        for (int c = 0; c < 4; ++c) {
            int col  = colBase + colW + c * 16 + l15;
            float bv = bias[col];
            #pragma unroll
            for (int r = 0; r < 8; ++r) {
                int lrow = rowW + g * 16 + half * 8 + r;
                int n    = rowBase + lrow;
                if (n < n_tokens) {
                    size_t idx = (size_t)n * H + col;
                    float val;
                    if (labs[lrow] != 0) val = acc[g][c][r] + bv;
                    else                 val = tokens[idx];
                    out[idx] = val;
                }
            }
        }
    }
}

// ---------------------------------------------------------------------------
// Fallback (no workspace): conversion of W in-kernel, single-buffered, 64-col
// ---------------------------------------------------------------------------
__global__ void __launch_bounds__(256)
fm_gemm_fallback(const float* __restrict__ tokens, const int* __restrict__ labels,
                 const float* __restrict__ lfeat, const float* __restrict__ W,
                 const float* __restrict__ bias, float* __restrict__ out, int n_tokens)
{
    __shared__ __align__(16) unsigned short fash[128 * BK];
    __shared__ __align__(16) unsigned short fbsh[64 * BK];
    __shared__ int labs[128];

    const int tid     = threadIdx.x;
    const int rowBase = blockIdx.y * 128;
    const int colBase = blockIdx.x * 64;

    if (tid < 128) {
        int n = rowBase + tid;
        labs[tid] = (n < n_tokens) ? labels[n] : 0;
    }
    __syncthreads();

    const int w    = tid >> 5;
    const int lane = tid & 31;
    const int half = lane >> 4;
    const int l15  = lane & 15;

    floatx8 acc[4] = {};

    for (int k0 = 0; k0 < K2; k0 += BK) {
        __syncthreads();
        #pragma unroll
        for (int it = 0; it < 4; ++it) {
            int t   = tid + it * 256;
            int row = t >> 3, ch = t & 7;
            int kc  = k0 + ch * 4;
            int n   = rowBase + row; if (n >= n_tokens) n = n_tokens - 1;
            const float* src;
            if (kc < H) src = tokens + (size_t)n * H + kc;
            else {
                int id = labs[row] - 1;
                if (id < 0) id = 0; if (id > NL - 1) id = NL - 1;
                src = lfeat + (size_t)id * H + (kc - H);
            }
            f4_t v = *(gf4p)src;
            uint2 pq; pq.x = f2bf_pk(v.x, v.y); pq.y = f2bf_pk(v.z, v.w);
            *(uint2*)&fash[row * BK + ch * 4] = pq;
        }
        #pragma unroll
        for (int it = 0; it < 2; ++it) {
            int t   = tid + it * 256;
            int col = t >> 3, ch = t & 7;
            const float* src = W + (size_t)(colBase + col) * K2 + k0 + ch * 4;
            f4_t v = *(gf4p)src;
            uint2 pq; pq.x = f2bf_pk(v.x, v.y); pq.y = f2bf_pk(v.z, v.w);
            *(uint2*)&fbsh[col * BK + ch * 4] = pq;
        }
        __syncthreads();

        bf16x16 af;
        {
            const char* ab = (const char*)fash;
            int row = w * 16 + l15;
            union { bf16x16 v; uint4 u[2]; } ua;
            ua.u[0] = *(const uint4*)(ab + row * 64 + half * 16);
            ua.u[1] = *(const uint4*)(ab + row * 64 + 32 + half * 16);
            af = ua.v;
        }
        #pragma unroll
        for (int c = 0; c < 4; ++c) {
            const char* bb = (const char*)fbsh;
            int col = c * 16 + l15;
            union { bf16x16 v; uint4 u[2]; } ubf;
            ubf.u[0] = *(const uint4*)(bb + col * 64 + half * 16);
            ubf.u[1] = *(const uint4*)(bb + col * 64 + 32 + half * 16);
            acc[c] = __builtin_amdgcn_wmma_f32_16x16x32_bf16(
                false, af, false, ubf.v, (short)0, acc[c], false, false);
        }
    }

    #pragma unroll
    for (int c = 0; c < 4; ++c) {
        int col  = colBase + c * 16 + l15;
        float bv = bias[col];
        #pragma unroll
        for (int r = 0; r < 8; ++r) {
            int lrow = w * 16 + half * 8 + r;
            int n    = rowBase + lrow;
            if (n < n_tokens) {
                size_t idx = (size_t)n * H + col;
                out[idx] = (labs[lrow] != 0) ? (acc[c][r] + bv) : tokens[idx];
            }
        }
    }
}

extern "C" void kernel_launch(void* const* d_in, const int* in_sizes, int n_in,
                              void* d_out, int out_size, void* d_ws, size_t ws_size,
                              hipStream_t stream) {
    const float* tokens = (const float*)d_in[0];   // com_features [B,S,H] f32
    const int*   labels = (const int*)  d_in[1];   // labels [B,S]
    const float* lfeat  = (const float*)d_in[2];   // label_feature [NL,H] f32
    const float* W      = (const float*)d_in[3];   // merge_w [H, 2H] f32
    const float* bias   = (const float*)d_in[4];   // merge_b [H] f32
    float*       out    = (float*)d_out;

    int n_tokens = in_sizes[0] / H;                // 65536

    const size_t wElems = (size_t)H * K2;          // 1,179,648
    if (ws_size >= wElems * sizeof(unsigned short)) {
        unsigned short* Wbf = (unsigned short*)d_ws;
        fm_convert_w<<<(int)(wElems / 1024), 256, 0, stream>>>(W, Wbf, (int)wElems);
        dim3 grid(H / BN, (n_tokens + BM - 1) / BM);   // (6, 512)
        fm_gemm_async<<<grid, 256, 0, stream>>>(tokens, labels, lfeat, Wbf, bias, out, n_tokens);
    } else {
        dim3 grid(H / 64, (n_tokens + 127) / 128);     // (12, 512)
        fm_gemm_fallback<<<grid, 256, 0, stream>>>(tokens, labels, lfeat, W, bias, out, n_tokens);
    }
}
